// Dinov3SelfAttention_2164663517462
// MI455X (gfx1250) — matrix operand order repaired and checked by reading, not compiled, not run on hardware
//
#include <hip/hip_runtime.h>

typedef _Float16 h8  __attribute__((ext_vector_type(8)));
typedef _Float16 h16 __attribute__((ext_vector_type(16)));
typedef float    f8  __attribute__((ext_vector_type(8)));

union HU { h16 v; h8 p[2]; };

#define WMMA16(a, b, c) \
  __builtin_amdgcn_wmma_f32_16x16x32_f16(false, (a), false, (b), (short)0, (c), false, false)

namespace {
constexpr int kB = 8, kS = 1374, kNH = 12, kD = 64, kHID = 768;
constexpr int kT = kB * kS;           // 10992 tokens (exactly 687 * 16)
constexpr int kSP = 1376;             // S padded to multiple of 32 for K-loop
constexpr int kPrefix = 5;            // S - N_ROT (cls + register tokens, no RoPE)
constexpr int kMT = kT / 16;          // 687 token tiles
constexpr int kNT = kHID / 64;        // 12 N-tiles of 64 == one head each
constexpr int kQT = (kS + 15) / 16;   // 86 q tiles
constexpr int kMG = (kMT + 7) / 8;    // 86 groups of 8 M-tiles per block
constexpr int kWS = 776;              // LDS row stride (768 + 8): bank-conflict-free
constexpr float kScale = 0.125f;      // HEAD_DIM^-0.5
constexpr float kNegInf = -1.0e30f;
} // namespace

// ---------------------------------------------------------------- f32 -> f16
__global__ void cvt_f16(const float* __restrict__ src, _Float16* __restrict__ dst, int n) {
  for (int i = blockIdx.x * blockDim.x + threadIdx.x; i < n; i += gridDim.x * blockDim.x)
    dst[i] = (_Float16)src[i];
}

// Zero the two padded K-positions of V^T so masked-tail WMMA contributions are exact.
__global__ void vt_pad(_Float16* __restrict__ Vt) {
  const int i = blockIdx.x * blockDim.x + threadIdx.x;
  constexpr int npad = kB * kNH * kD * (kSP - kS);
  if (i < npad) {
    const int row = i >> 1;            // (b*NH+h)*D + d
    const int s = kS + (i & 1);
    Vt[(size_t)row * kSP + s] = (_Float16)0.f;
  }
}

// ------------------------------------------------------- GEMM C = A * W^T (+b)
// Block = 8 waves = 8 M-tiles (128 tokens) x one 64-channel N-tile (one head).
// The W slice for the N-tile (64 x 768 f16 = 96KB) is staged into LDS once via
// CDNA5 async-to-LDS loads and shared by all 8 waves; B fragments then come
// from LDS (conflict-free thanks to the 776-half row stride), A fragments
// stream from global.  RoPE pairs (d, d+-32) stay in-lane (64-wide N tile).
// mode 0: Q (bias, RoPE, *0.125 -> [B,NH,S,D])   mode 1: K (RoPE, no bias)
// mode 2: V (bias, -> [B,NH,D,SP] transposed)    mode 3: out proj (bias, f32)
__global__ __launch_bounds__(256) void gemm_ep(
    const _Float16* __restrict__ A, const _Float16* __restrict__ W,
    const float* __restrict__ bias,
    const float* __restrict__ sinp, const float* __restrict__ cosp,
    _Float16* __restrict__ outH, float* __restrict__ outF, int mode) {
  __shared__ _Float16 wlds[64 * kWS];  // 99,328 bytes

  const int nt = blockIdx.x % kNT;
  const int mg = blockIdx.x / kNT;
  const int n0 = nt * 64;
  const int lane = threadIdx.x & 31;
  const int hl = lane & 15, hi = lane >> 4;

  // ---- cooperative async stage of W[n0:n0+64, 0:768] into LDS --------------
  {
    const int tid = threadIdx.x;
#pragma unroll
    for (int it = 0; it < 24; ++it) {               // 6144 b128 chunks / 256 thr
      const int c = tid + it * 256;
      const int row = c / 96, col = (c % 96) * 8;   // 96 chunks per 768-wide row
      const unsigned goff = (unsigned)(((n0 + row) * kHID + col) * 2);
      const unsigned loff = (unsigned)(uintptr_t)&wlds[row * kWS + col];
      asm volatile("global_load_async_to_lds_b128 %0, %1, %2"
                   :: "v"(loff), "v"(goff), "s"(W)
                   : "memory");
    }
    asm volatile("s_wait_asynccnt 0x0" ::: "memory");
  }
  __syncthreads();

  // ---- per-wave 16x64 WMMA microkernel -------------------------------------
  const int mtile = mg * 8 + (threadIdx.x >> 5);
  const int mtc = mtile < kMT ? mtile : kMT - 1;    // clamp ragged group
  const int m0 = mtc * 16;

  f8 acc[4];
  for (int j = 0; j < 4; ++j)
    for (int e = 0; e < 8; ++e) acc[j][e] = 0.f;

  const _Float16* arow = A + (size_t)(m0 + hl) * kHID;
  for (int k0 = 0; k0 < kHID; k0 += 32) {
    HU a;
    a.p[0] = *(const h8*)(arow + k0 + hi * 16);
    a.p[1] = *(const h8*)(arow + k0 + hi * 16 + 8);
#pragma unroll
    for (int j = 0; j < 4; ++j) {
      HU bf;
      const _Float16* wp = &wlds[(16 * j + hl) * kWS + k0 + hi * 16];
      bf.p[0] = *(const h8*)wp;
      bf.p[1] = *(const h8*)(wp + 8);
      acc[j] = WMMA16(a.v, bf.v, acc[j]);
    }
  }

  if (mtile >= kMT) return;  // wave-uniform; only stores are skipped

  float bcol[4];
#pragma unroll
  for (int j = 0; j < 4; ++j) bcol[j] = bias ? bias[n0 + 16 * j + hl] : 0.f;

#pragma unroll
  for (int i = 0; i < 8; ++i) {
    const int t = m0 + hi * 8 + i;  // C layout: lanes<16 rows 0..7, lanes>=16 rows 8..15
    float x[4];
#pragma unroll
    for (int j = 0; j < 4; ++j) x[j] = acc[j][i] + bcol[j];

    if (mode == 3) {
      float* op = outF + (size_t)t * kHID + n0 + hl;
#pragma unroll
      for (int j = 0; j < 4; ++j) op[16 * j] = x[j];
    } else {
      const int b = t / kS, s = t % kS, h = nt;
      if (mode == 2) {  // V^T store
        _Float16* op = outH + ((size_t)(b * kNH + h) * kD + hl) * kSP + s;
#pragma unroll
        for (int j = 0; j < 4; ++j) op[(size_t)(16 * j) * kSP] = (_Float16)x[j];
      } else {          // Q / K with RoPE on patch tokens
        float y[4];
        if (s >= kPrefix) {
          const float* sr = sinp + (size_t)(s - kPrefix) * kD;
          const float* cr = cosp + (size_t)(s - kPrefix) * kD;
#pragma unroll
          for (int j = 0; j < 4; ++j) {
            const int d = 16 * j + hl;
            const float rot = (j < 2) ? -x[j + 2] : x[j - 2];  // rotate_half
            y[j] = x[j] * cr[d] + rot * sr[d];
          }
        } else {
#pragma unroll
          for (int j = 0; j < 4; ++j) y[j] = x[j];
        }
        _Float16* op = outH + ((size_t)(b * kNH + h) * kS + s) * kD + hl;
#pragma unroll
        for (int j = 0; j < 4; ++j)
          op[16 * j] = (_Float16)(mode == 0 ? y[j] * kScale : y[j]);
      }
    }
  }
}

// ------------------------------------------------ flash attention, one wave per
// (b, h, 16-query tile).  Computes S^T = K * Q^T so the probability tile lands
// directly in WMMA A-fragment layout for the P*V product (no LDS transpose).
__global__ __launch_bounds__(256) void attn(
    const _Float16* __restrict__ Q, const _Float16* __restrict__ K,
    const _Float16* __restrict__ Vt, _Float16* __restrict__ ctx) {
  const int lane = threadIdx.x & 31;
  const int wid = blockIdx.x * (blockDim.x >> 5) + (threadIdx.x >> 5);
  if (wid >= kB * kNH * kQT) return;
  const int qt = wid % kQT;
  const int bh = wid / kQT;
  const int b = bh / kNH, h = bh % kNH;
  const int q0 = qt * 16;
  const int hl = lane & 15, hi = lane >> 4;

  const _Float16* Qb = Q + (size_t)bh * kS * kD;
  const _Float16* Kb = K + (size_t)bh * kS * kD;
  const _Float16* Vb = Vt + (size_t)bh * kD * kSP;

  // Q as WMMA-B fragments (lane = query column, contiguous d), kept resident.
  int qc = q0 + hl;
  if (qc >= kS) qc = kS - 1;  // clamped dup rows; their outputs are never stored
  const h16 qf0 = *(const h16*)(Qb + (size_t)qc * kD + hi * 16);
  const h16 qf1 = *(const h16*)(Qb + (size_t)qc * kD + 32 + hi * 16);

  f8 acc[4];
  for (int j = 0; j < 4; ++j)
    for (int e = 0; e < 8; ++e) acc[j][e] = 0.f;
  float mrun = kNegInf, lrun = 0.f;

  for (int kt = 0; kt < kSP; kt += 32) {
    if (kt + 32 < kSP) {  // prefetch next K/V chunk (global_prefetch_b8)
      __builtin_prefetch(Kb + (size_t)(kt + 32 + hl) * kD, 0, 1);
      __builtin_prefetch(Kb + (size_t)(kt + 48 + hl) * kD, 0, 1);
      __builtin_prefetch(Vb + (size_t)hl * kSP + kt + 32, 0, 1);
    }
    int r0 = kt + hl;       if (r0 >= kS) r0 = kS - 1;
    int r1 = kt + 16 + hl;  if (r1 >= kS) r1 = kS - 1;
    f8 s0, s1;
    for (int e = 0; e < 8; ++e) { s0[e] = 0.f; s1[e] = 0.f; }
    {
      HU ka, kb;
      const _Float16* kp0 = Kb + (size_t)r0 * kD + hi * 16;
      const _Float16* kp1 = Kb + (size_t)r1 * kD + hi * 16;
      ka.p[0] = *(const h8*)kp0;        ka.p[1] = *(const h8*)(kp0 + 8);
      kb.p[0] = *(const h8*)kp1;        kb.p[1] = *(const h8*)(kp1 + 8);
      s0 = WMMA16(ka.v, qf0, s0);
      s1 = WMMA16(kb.v, qf0, s1);
      ka.p[0] = *(const h8*)(kp0 + 32); ka.p[1] = *(const h8*)(kp0 + 40);
      kb.p[0] = *(const h8*)(kp1 + 32); kb.p[1] = *(const h8*)(kp1 + 40);
      s0 = WMMA16(ka.v, qf1, s0);
      s1 = WMMA16(kb.v, qf1, s1);
    }
    if (kt + 32 > kS) {  // mask K tail (uniform branch)
#pragma unroll
      for (int e = 0; e < 8; ++e) {
        if (kt + hi * 8 + e >= kS)      s0[e] = kNegInf;
        if (kt + 16 + hi * 8 + e >= kS) s1[e] = kNegInf;
      }
    }
    // online softmax over k: per-q values split across the (lane, lane^16) pair
    float tm = kNegInf;
#pragma unroll
    for (int e = 0; e < 8; ++e) tm = fmaxf(tm, fmaxf(s0[e], s1[e]));
    tm = fmaxf(tm, __shfl_xor(tm, 16, 32));
    const float mnew = fmaxf(mrun, tm);
    const float fac = __expf(mrun - mnew);
    mrun = mnew;
    float ps = 0.f;
    HU pa;  // S^T C-layout == A-fragment layout for P: direct repack in-lane
#pragma unroll
    for (int e = 0; e < 8; ++e) {
      const float p0 = __expf(s0[e] - mnew);
      const float p1 = __expf(s1[e] - mnew);
      ps += p0 + p1;
      pa.v[e] = (_Float16)p0;
      pa.v[8 + e] = (_Float16)p1;
    }
    ps += __shfl_xor(ps, 16, 32);
    lrun = lrun * fac + ps;
    // rescale running context: broadcast per-row factor (row q lives at lane q)
    float facr[8];
#pragma unroll
    for (int e = 0; e < 8; ++e) facr[e] = __shfl(fac, hi * 8 + e, 32);
#pragma unroll
    for (int j = 0; j < 4; ++j)
#pragma unroll
      for (int e = 0; e < 8; ++e) acc[j][e] *= facr[e];
    // ctx += P * V   (V^T rows give contiguous B fragments)
#pragma unroll
    for (int j = 0; j < 4; ++j) {
      HU vf;
      const _Float16* vp = Vb + (size_t)(16 * j + hl) * kSP + kt + hi * 8;
      vf.p[0] = *(const h8*)vp;
      vf.p[1] = *(const h8*)(vp + 16);
      acc[j] = WMMA16(pa.v, vf.v, acc[j]);
    }
  }

  float linv[8];
#pragma unroll
  for (int e = 0; e < 8; ++e) linv[e] = 1.f / __shfl(lrun, hi * 8 + e, 32);

#pragma unroll
  for (int e = 0; e < 8; ++e) {
    const int s = q0 + hi * 8 + e;
    if (s < kS) {
      _Float16* cp = ctx + ((size_t)b * kS + s) * kHID + h * kD + hl;
#pragma unroll
      for (int j = 0; j < 4; ++j) cp[16 * j] = (_Float16)(acc[j][e] * linv[e]);
    }
  }
}

// ---------------------------------------------------------------------- driver
extern "C" void kernel_launch(void* const* d_in, const int* in_sizes, int n_in,
                              void* d_out, int out_size, void* d_ws, size_t ws_size,
                              hipStream_t stream) {
  const float* hs   = (const float*)d_in[0];
  const float* sinp = (const float*)d_in[1];
  const float* cosp = (const float*)d_in[2];
  const float* Wq   = (const float*)d_in[3];
  const float* bq   = (const float*)d_in[4];
  const float* Wk   = (const float*)d_in[5];
  const float* Wv   = (const float*)d_in[6];
  const float* bv   = (const float*)d_in[7];
  const float* Wp   = (const float*)d_in[8];
  const float* bp   = (const float*)d_in[9];
  (void)in_sizes; (void)n_in; (void)out_size; (void)ws_size;

  char* ws = (char*)d_ws;
  size_t off = 0;
  auto take = [&](size_t bytes) -> char* {
    char* p = ws + off;
    off += (bytes + 255) & ~(size_t)255;
    return p;
  };
  _Float16* hsh = (_Float16*)take((size_t)kT * kHID * 2);
  _Float16* wqh = (_Float16*)take((size_t)kHID * kHID * 2);
  _Float16* wkh = (_Float16*)take((size_t)kHID * kHID * 2);
  _Float16* wvh = (_Float16*)take((size_t)kHID * kHID * 2);
  _Float16* wph = (_Float16*)take((size_t)kHID * kHID * 2);
  _Float16* Qh  = (_Float16*)take((size_t)kB * kNH * kS * kD * 2);
  _Float16* Kh  = (_Float16*)take((size_t)kB * kNH * kS * kD * 2);
  _Float16* Vt  = (_Float16*)take((size_t)kB * kNH * kD * kSP * 2);
  _Float16* ctx = (_Float16*)take((size_t)kT * kHID * 2);

  const int nHS = kT * kHID, nW = kHID * kHID;
  cvt_f16<<<2048, 256, 0, stream>>>(hs, hsh, nHS);
  cvt_f16<<<1024, 256, 0, stream>>>(Wq, wqh, nW);
  cvt_f16<<<1024, 256, 0, stream>>>(Wk, wkh, nW);
  cvt_f16<<<1024, 256, 0, stream>>>(Wv, wvh, nW);
  cvt_f16<<<1024, 256, 0, stream>>>(Wp, wph, nW);

  const int gemmBlocks = kMG * kNT;  // 86 * 12 = 1032
  gemm_ep<<<gemmBlocks, 256, 0, stream>>>(hsh, wqh, bq, sinp, cosp, Qh, nullptr, 0);
  gemm_ep<<<gemmBlocks, 256, 0, stream>>>(hsh, wkh, nullptr, sinp, cosp, Kh, nullptr, 1);
  gemm_ep<<<gemmBlocks, 256, 0, stream>>>(hsh, wvh, bv, sinp, cosp, Vt, nullptr, 2);
  vt_pad<<<48, 256, 0, stream>>>(Vt);

  const int attnBlocks = (kB * kNH * kQT + 7) / 8;
  attn<<<attnBlocks, 256, 0, stream>>>(Qh, Kh, Vt, ctx);

  gemm_ep<<<gemmBlocks, 256, 0, stream>>>(ctx, wph, bp, sinp, cosp, nullptr, (float*)d_out, 3);
}
